// BiLSTM_20040317403332
// MI455X (gfx1250) — matrix operand order repaired and checked by reading, not compiled, not run on hardware
//
#include <hip/hip_runtime.h>
#include <hip/hip_bf16.h>
#include <stdint.h>

#define VOCAB  32000
#define EMBED  512
#define HIDDEN 512
#define BATCH  8
#define SEQ    512
#define GATES  2048   // 4*HIDDEN

typedef __attribute__((ext_vector_type(16))) __bf16        v16bf;
typedef __attribute__((ext_vector_type(8)))  float         v8f;
typedef __attribute__((ext_vector_type(8)))  unsigned short us8;
typedef __attribute__((ext_vector_type(16))) unsigned short us16;

static __device__ __forceinline__ unsigned short f2bf(float f) {
    unsigned u = __float_as_uint(f);
    u += 0x7FFFu + ((u >> 16) & 1u);          // round-to-nearest-even
    return (unsigned short)(u >> 16);
}
static __device__ __forceinline__ float sigm(float x) {
    return 1.0f / (1.0f + __expf(-x));
}
static __device__ __forceinline__ float tanh_fast(float x) {
    float e = __expf(2.0f * x);
    return (e - 1.0f) / (e + 1.0f);
}

union Frag16 { v16bf v; us8 h[2]; us16 u; };

// ---------------------------------------------------------------------------
// C[M,N] = A[M,K] (bf16 row-major) * B_w[N,K] (bf16 row-major, used as B^T)
//          + bias[N]   (fp32 accumulate via v_wmma_f32_16x16x32_bf16)
// One wave computes a 16x64 C tile (4 WMMA accumulators). Requires M%16==0,
// N%64==0, K%32==0. Launch exactly ceil(tiles/8) blocks of 256 threads.
// ---------------------------------------------------------------------------
__global__ void gemm_bf16_wmma(const unsigned short* __restrict__ A,
                               const unsigned short* __restrict__ Bw,
                               const float* __restrict__ bias,
                               float* __restrict__ C,
                               int M, int N, int K)
{
    const int lane   = threadIdx.x & 31;
    const int wid    = blockIdx.x * (blockDim.x >> 5) + (threadIdx.x >> 5);
    const int tilesN = N >> 6;
    const int tiles  = (M >> 4) * tilesN;
    if (wid >= tiles) return;

    const int tm   = wid / tilesN;
    const int tn   = wid % tilesN;
    const int l15  = lane & 15;
    const int half = lane >> 4;

    // A-matrix 16x32 bf16 layout: lane holds row (lane&15);
    //   elems 0..7 -> K = kb + 8*half + e, elems 8..15 -> K = kb + 16 + 8*half + (e-8)
    const unsigned short* aRow = A + (size_t)(tm * 16 + l15) * K;
    // B-matrix 32x16 bf16 layout: lane holds column (lane&15);
    //   elems 0..7 -> K = kb + 8*half + e, elems 8..15 -> K = kb + 16 + 8*half + (e-8)
    const unsigned short* bRow[4];
#pragma unroll
    for (int j = 0; j < 4; ++j)
        bRow[j] = Bw + (size_t)(tn * 64 + j * 16 + l15) * K;

    v8f acc[4] = {};

#pragma unroll 4
    for (int kb = 0; kb < K; kb += 32) {
        Frag16 af;
        af.h[0] = *(const us8*)(aRow + kb +      8 * half);
        af.h[1] = *(const us8*)(aRow + kb + 16 + 8 * half);
#pragma unroll
        for (int j = 0; j < 4; ++j) {
            Frag16 bf;
            bf.h[0] = *(const us8*)(bRow[j] + kb +      8 * half);
            bf.h[1] = *(const us8*)(bRow[j] + kb + 16 + 8 * half);
            acc[j] = __builtin_amdgcn_wmma_f32_16x16x32_bf16(
                false, af.v, false, bf.v, (short)0, acc[j], false, false);
        }
    }

    // C/D layout: VGPR r, lane l -> row = r + 8*(l>>4), col = l&15
#pragma unroll
    for (int j = 0; j < 4; ++j) {
        const size_t col = (size_t)tn * 64 + j * 16 + l15;
        const float  bv  = bias ? bias[col] : 0.0f;
#pragma unroll
        for (int r = 0; r < 8; ++r) {
            const size_t row = (size_t)tm * 16 + r + 8 * half;
            C[row * (size_t)N + col] = acc[j][r] + bv;
        }
    }
}

// ---------------------------------------------------------------------------
// Embedding gather + fp32 -> bf16, into x[(t*BATCH+b)*EMBED + e]
// ---------------------------------------------------------------------------
__global__ void gather_embed_bf16(const int* __restrict__ idx,
                                  const float* __restrict__ emb,
                                  unsigned short* __restrict__ xbf)
{
    int g = blockIdx.x * blockDim.x + threadIdx.x;   // < SEQ*BATCH*EMBED
    int e   = g & (EMBED - 1);
    int row = g >> 9;                 // / EMBED
    int b   = row & (BATCH - 1);
    int t   = row >> 3;               // / BATCH
    int tok = idx[b * SEQ + t];
    xbf[g] = f2bf(emb[(size_t)tok * EMBED + e]);
}

__global__ void f32_to_bf16(const float* __restrict__ in,
                            unsigned short* __restrict__ out, int n)
{
    int g = blockIdx.x * blockDim.x + threadIdx.x;
    if (g < n) out[g] = f2bf(in[g]);
}

// zero c (BATCH*HIDDEN fp32) and h_bf (16*HIDDEN bf16, rows 8..15 stay zero)
__global__ void init_state(float* __restrict__ c, unsigned short* __restrict__ h_bf)
{
    int g = blockIdx.x * blockDim.x + threadIdx.x;   // 16*HIDDEN threads
    if (g < BATCH * HIDDEN) c[g] = 0.0f;
    if (g < 16 * HIDDEN)    h_bf[g] = 0;
}

// ---------------------------------------------------------------------------
// Elementwise LSTM cell update for one timestep.
// gates: 16 x GATES fp32 (rows 0..7 valid) = h_prev @ W_hh^T + b_hh
// xg_t:  BATCH x GATES fp32 slice of precomputed x @ W_ih^T + b_ih
// Writes c (fp32 state), h_bf (16 x HIDDEN bf16, next-step A matrix),
// hs_bf[(b*SEQ + t)*HIDDEN + j] (bf16, A matrix of the final GEMM).
// ---------------------------------------------------------------------------
__global__ void lstm_step(const float* __restrict__ gates,
                          const float* __restrict__ xg_t,
                          float* __restrict__ c,
                          unsigned short* __restrict__ h_bf,
                          unsigned short* __restrict__ hs_bf,
                          int t)
{
    int g = blockIdx.x * blockDim.x + threadIdx.x;   // < BATCH*HIDDEN
    int b = g >> 9;                  // / HIDDEN
    int j = g & (HIDDEN - 1);
    const float* gr = gates + (size_t)b * GATES;
    const float* xr = xg_t  + (size_t)b * GATES;

    float ig = sigm     (gr[j]                 + xr[j]);
    float fg = sigm     (gr[HIDDEN + j]        + xr[HIDDEN + j]);
    float gg = tanh_fast(gr[2 * HIDDEN + j]    + xr[2 * HIDDEN + j]);
    float og = sigm     (gr[3 * HIDDEN + j]    + xr[3 * HIDDEN + j]);

    float cn = fg * c[g] + ig * gg;
    c[g] = cn;
    float h = og * tanh_fast(cn);

    unsigned short hb = f2bf(h);
    h_bf[(size_t)b * HIDDEN + j] = hb;                       // rows 8..15 remain 0
    hs_bf[((size_t)b * SEQ + t) * HIDDEN + j] = hb;          // (B,T,H) layout
}

// ---------------------------------------------------------------------------
extern "C" void kernel_launch(void* const* d_in, const int* in_sizes, int n_in,
                              void* d_out, int out_size, void* d_ws, size_t ws_size,
                              hipStream_t stream)
{
    (void)in_sizes; (void)n_in; (void)out_size; (void)ws_size;

    const int*   input = (const int*)  d_in[0];
    const float* emb   = (const float*)d_in[1];
    const float* W_ih  = (const float*)d_in[2];
    const float* b_ih  = (const float*)d_in[3];
    const float* W_hh  = (const float*)d_in[4];
    const float* b_hh  = (const float*)d_in[5];
    // d_in[6..9]: backward-direction weights -- dead code in the reference (DCE'd)
    const float* W_out = (const float*)d_in[10];
    const float* b_out = (const float*)d_in[11];
    float* out = (float*)d_out;

    char* ws = (char*)d_ws;
    size_t off = 0;
    auto alloc = [&](size_t bytes) -> void* {
        void* p = ws + off;
        off = (off + bytes + 255) & ~(size_t)255;
        return p;
    };

    unsigned short* Wih_bf  = (unsigned short*)alloc((size_t)GATES * EMBED * 2);   //  2 MB
    unsigned short* Whh_bf  = (unsigned short*)alloc((size_t)GATES * HIDDEN * 2);  //  2 MB
    unsigned short* Wout_bf = (unsigned short*)alloc((size_t)VOCAB * HIDDEN * 2);  // 32.8 MB
    unsigned short* x_bf    = (unsigned short*)alloc((size_t)SEQ * BATCH * EMBED * 2);   // 4 MB
    float*          xg      = (float*)         alloc((size_t)SEQ * BATCH * GATES * 4);   // 33.5 MB
    unsigned short* h_bf    = (unsigned short*)alloc((size_t)16 * HIDDEN * 2);
    float*          c_state = (float*)         alloc((size_t)BATCH * HIDDEN * 4);
    float*          gates   = (float*)         alloc((size_t)16 * GATES * 4);
    unsigned short* hs_bf   = (unsigned short*)alloc((size_t)BATCH * SEQ * HIDDEN * 2);  // 4 MB

    const int T = 256; // threads per block (8 waves)

    // 1) weight conversions fp32 -> bf16 (resident in L2 afterwards)
    {
        int n;
        n = GATES * EMBED;   f32_to_bf16<<<(n + T - 1) / T, T, 0, stream>>>(W_ih,  Wih_bf,  n);
        n = GATES * HIDDEN;  f32_to_bf16<<<(n + T - 1) / T, T, 0, stream>>>(W_hh,  Whh_bf,  n);
        n = VOCAB * HIDDEN;  f32_to_bf16<<<(n + T - 1) / T, T, 0, stream>>>(W_out, Wout_bf, n);
    }

    // 2) zero the recurrent state (every call: graph-replay safe)
    init_state<<<(16 * HIDDEN + T - 1) / T, T, 0, stream>>>(c_state, h_bf);

    // 3) embedding gather -> bf16 x in (T*B, E) layout
    {
        int n = SEQ * BATCH * EMBED;
        gather_embed_bf16<<<(n + T - 1) / T, T, 0, stream>>>(input, emb, x_bf);
    }

    // 4) input projection: xg[T*B, 4H] = x @ W_ih^T + b_ih  (one big WMMA GEMM)
    {
        int tiles = (SEQ * BATCH / 16) * (GATES / 64);       // 8192 wave tiles
        gemm_bf16_wmma<<<tiles / 8, T, 0, stream>>>(x_bf, Wih_bf, b_ih, xg,
                                                    SEQ * BATCH, GATES, EMBED);
    }

    // 5) sequential LSTM scan: per step a 32-wave WMMA GEMM + elementwise cell
    for (int t = 0; t < SEQ; ++t) {
        // gates[16, 4H] = h_prev @ W_hh^T + b_hh   (M padded 8->16 with zeros)
        gemm_bf16_wmma<<<4, T, 0, stream>>>(h_bf, Whh_bf, b_hh, gates,
                                            16, GATES, HIDDEN);
        lstm_step<<<(BATCH * HIDDEN) / T, T, 0, stream>>>(
            gates, xg + (size_t)t * BATCH * GATES, c_state, h_bf, hs_bf, t);
    }

    // 6) output projection: out[B*T, V] = hs @ W_out^T + b_out (134 GFLOP,
    //    write-bandwidth bound: 524 MB of fp32 logits)
    {
        int tiles = (BATCH * SEQ / 16) * (VOCAB / 64);       // 128000 wave tiles
        gemm_bf16_wmma<<<tiles / 8, T, 0, stream>>>(hs_bf, Wout_bf, b_out, out,
                                                    BATCH * SEQ, VOCAB, HIDDEN);
    }
}
